// MoELayer_73143293051461
// MI455X (gfx1250) — matrix operand order repaired and checked by reading, not compiled, not run on hardware
//
#include <hip/hip_runtime.h>

// CDNA5 WMMA fragment types (wave32)
typedef __attribute__((ext_vector_type(16))) __bf16        v16bf;
typedef __attribute__((ext_vector_type(8)))  float         v8f;
typedef __attribute__((ext_vector_type(8)))  unsigned int  v8u;
typedef __attribute__((ext_vector_type(4)))  unsigned int  v4u;
typedef __attribute__((ext_vector_type(2)))  float         f32x2;
typedef __attribute__((ext_vector_type(2)))  __bf16        bf16x2;

#define CIN    256
#define COUT   256
#define HH     64
#define WW     64
#define KTOT   2304          // CIN * 9, K ordered tap-major: k = (kh*3+kw)*256 + ci
#define KC     32
#define NSTAGE 72            // KTOT / KC ; 8 chunks per tap, 9 taps
#define LDSS   40            // LDS row stride in halfwords: 80B rows, 16B-aligned frags

struct U8 { v4u a, b; };     // 32B fragment container

// pack two floats into one dword of bf16 via packed converter (RNE)
__device__ __forceinline__ unsigned int pack2_bf16(float a, float b) {
    f32x2 f; f.x = a; f.y = b;
    bf16x2 h = __builtin_convertvector(f, bf16x2);   // -> v_cvt_pk_bf16_f32
    return __builtin_bit_cast(unsigned int, h);
}

// ---------------------------------------------------------------------------
// One-time weight transform: fp32 OIHW -> bf16 prepacked in the exact WMMA
// A-fragment lane layout, tap-major K.  Layout (uints):
//   Aprep[ ((chunk*16 + mtile)*32 + lane)*8 + p ]
// ---------------------------------------------------------------------------
__global__ __launch_bounds__(256)
void prep_weights(const float* __restrict__ W, unsigned int* __restrict__ Aprep)
{
    const int idx = blockIdx.x * 256 + threadIdx.x;   // 72*16*32*8 = 294912
    const int p  = idx & 7;
    const int l  = (idx >> 3) & 31;
    const int t  = (idx >> 8) & 15;
    const int c  = idx >> 12;                          // chunk 0..71
    const int m  = t * 16 + (l & 15);                  // cout row
    const int hi = l >> 4;
    const int j  = 2 * p;
    const int kf = j + 16 * hi;                        // chunk-local K (even)
    const int k0 = c * KC + kf;                        // tap-major linear K
    const int r0 = k0 >> 8, ci0 = k0 & 255;
    const int r1 = (k0 + 1) >> 8, ci1 = (k0 + 1) & 255;
    const float f0 = W[m * KTOT + ci0 * 9 + r0];
    const float f1 = W[m * KTOT + ci1 * 9 + r1];
    Aprep[idx] = pack2_bf16(f0, f1);
}

// ---------------------------------------------------------------------------
// Main implicit-GEMM conv: M=COUT, N=B*H*W, K=CIN*9 (tap-major).
// WG tile 256(cout) x 128(2 h-rows x 64 w), 8 waves, wave tile 64x64
// (16 wmma / stage / wave).  B double-buffered in LDS with software
// prefetch; A fragments prepacked in WMMA lane layout, streamed from L2.
// ---------------------------------------------------------------------------
__global__ __launch_bounds__(256)
void conv3x3_gate_wmma(const float* __restrict__ x,
                       const float* __restrict__ gate,
                       const float* __restrict__ bias,
                       const unsigned int* __restrict__ Aprep,
                       float* __restrict__ out)
{
    __shared__ __align__(16) unsigned short Bs[2][128 * LDSS];   // 2 x 10240 B

    const int tid    = threadIdx.x;
    const int lane   = tid & 31;
    const int wave   = tid >> 5;
    const int laneHi = lane >> 4;
    const int lane16 = lane & 15;

    const int nTile = blockIdx.x;          // b*32 + hTile
    const int b  = nTile >> 5;
    const int hT = nTile & 31;             // h base = hT*2

    const int mWave = (wave & 3) * 64;     // 4 M groups of 64 couts
    const int nWave = (wave >> 2) * 64;    // 2 N groups of 64 spatial

    const float* __restrict__ xb = x + (size_t)b * (CIN * HH * WW);

    // B staging: thread owns spatial point nB and 8 consecutive K-pairs
    // jp = jp0*8 + it  ->  kk = 2*jp, ci = ciBase + jp0*16 + 2*it (+1)
    const int nB  = tid & 127;
    const int jp0 = tid >> 7;              // 0..1
    const int hh  = nB >> 6;
    const int wb  = nB & 63;

    // A fragment base for this wave's 4 m-subtiles
    const int tA = mWave >> 4;             // (wave&3)*4
    const unsigned int* aBase = Aprep + ((size_t)tA * 32 + lane) * 8;

    v8f acc[4][4] = {};

    // stage-uniform gather base for chunk s (tap fixed per chunk)
    auto stage_addr = [&](int s, bool& valid, const float*& src) {
        const int r   = s >> 3;            // tap 0..8
        const int kh  = r / 3;
        const int kw  = r - 3 * kh;
        const int ciB = (s & 7) * KC + jp0 * 16;
        const int h   = hT * 2 + hh + kh - 1;
        const int wsx = wb + kw - 1;
        valid = ((unsigned)h < (unsigned)HH) & ((unsigned)wsx < (unsigned)WW);
        const int off = valid ? (h * WW + wsx) : 0;   // clamped, always in-bounds
        src = xb + (size_t)ciB * (HH * WW) + off;
    };

    // convert + pack + zero-invalid + 2x ds_store_b128 into buffer `dst`
    auto stage_store = [&](int dst, bool valid, const float (&p0)[8], const float (&p1)[8]) {
        #pragma unroll
        for (int g = 0; g < 2; ++g) {
            v4u pk;
            #pragma unroll
            for (int e = 0; e < 4; ++e) {
                unsigned int u = pack2_bf16(p0[4 * g + e], p1[4 * g + e]);
                pk[e] = valid ? u : 0u;
            }
            *(v4u*)&Bs[dst][nB * LDSS + jp0 * 16 + g * 8] = pk;
        }
    };

    // ---- prologue: stage chunk 0 into Bs[0] ----
    {
        bool valid; const float* src;
        stage_addr(0, valid, src);
        float p0[8], p1[8];
        #pragma unroll
        for (int it = 0; it < 8; ++it) {
            p0[it] = src[it * 2 * (HH * WW)];
            p1[it] = src[it * 2 * (HH * WW) + (HH * WW)];
        }
        stage_store(0, valid, p0, p1);
    }
    __syncthreads();

    for (int s = 0; s < NSTAGE; ++s) {
        const int buf = s & 1;

        // ---- issue next chunk's global loads (overlap with WMMA below) ----
        float pv0[8], pv1[8];
        bool pvalid = false;
        if (s + 1 < NSTAGE) {
            const float* psrc;
            stage_addr(s + 1, pvalid, psrc);
            #pragma unroll
            for (int it = 0; it < 8; ++it) {
                pv0[it] = psrc[it * 2 * (HH * WW)];
                pv1[it] = psrc[it * 2 * (HH * WW) + (HH * WW)];
            }
        }

        // ---- B fragments: 2x ds_load_b128 each (32B contiguous, 16B aligned) ----
        v16bf bfrag[4];
        #pragma unroll
        for (int ni = 0; ni < 4; ++ni) {
            const unsigned short* base =
                &Bs[buf][(nWave + ni * 16 + lane16) * LDSS + 16 * laneHi];
            U8 t;
            t.a = *(const v4u*)(base);
            t.b = *(const v4u*)(base + 8);
            bfrag[ni] = __builtin_bit_cast(v16bf, t);
        }

        // ---- 16x v_wmma_f32_16x16x32_bf16 (A prepacked, 32B load per mi) ----
        #pragma unroll
        for (int mi = 0; mi < 4; ++mi) {
            const v8u au = *(const v8u*)(aBase + ((size_t)s * 16 + mi) * 256);
            const v16bf afrag = __builtin_bit_cast(v16bf, au);
            #pragma unroll
            for (int ni = 0; ni < 4; ++ni)
                acc[mi][ni] = __builtin_amdgcn_wmma_f32_16x16x32_bf16(
                    false, afrag, false, bfrag[ni],
                    (short)0, acc[mi][ni], false, false);
        }

        // ---- convert + store prefetched chunk into the other buffer ----
        if (s + 1 < NSTAGE)
            stage_store(buf ^ 1, pvalid, pv0, pv1);
        __syncthreads();
    }

    // ---- epilogue: (acc + bias[co]) * gate[b,co], f32 NCHW stores ----
    #pragma unroll
    for (int mi = 0; mi < 4; ++mi) {
        #pragma unroll
        for (int ni = 0; ni < 4; ++ni) {
            const int nLoc = nWave + ni * 16 + lane16;
            const int h = hT * 2 + (nLoc >> 6);
            const int w = nLoc & 63;
            #pragma unroll
            for (int vi = 0; vi < 8; ++vi) {
                const int co = mWave + mi * 16 + vi + 8 * laneHi;   // ni-independent -> CSE'd loads
                const float r = (acc[mi][ni][vi] + bias[co]) * gate[b * COUT + co];
                out[(((size_t)b * COUT + co) << 12) + (h << 6) + w] = r;
            }
        }
    }
}

extern "C" void kernel_launch(void* const* d_in, const int* in_sizes, int n_in,
                              void* d_out, int out_size, void* d_ws, size_t ws_size,
                              hipStream_t stream) {
    (void)in_sizes; (void)n_in; (void)out_size; (void)ws_size;
    const float* x    = (const float*)d_in[0];  // (32,256,64,64)
    const float* gate = (const float*)d_in[1];  // (32,256)
    const float* Wt   = (const float*)d_in[2];  // (256,256,3,3)
    const float* bias = (const float*)d_in[3];  // (256,)
    float* out        = (float*)d_out;          // (32,256,64,64) f32

    unsigned int* Aprep = (unsigned int*)d_ws;  // 1,179,648 B prepacked weights

    prep_weights<<<1152, 256, 0, stream>>>(Wt, Aprep);
    conv3x3_gate_wmma<<<dim3(1024, 1, 1), 256, 0, stream>>>(x, gate, bias, Aprep, out);
}
